// CausalSelfAttention2_71640054497491
// MI455X (gfx1250) — matrix operand order repaired and checked by reading, not compiled, not run on hardware
//
#include <hip/hip_runtime.h>

// ---------------- types ----------------
typedef __bf16 bf16_t;
typedef __attribute__((ext_vector_type(16))) __bf16 v16bf;
typedef __attribute__((ext_vector_type(8)))  __bf16 v8bf;
typedef __attribute__((ext_vector_type(4)))  __bf16 v4bf;
typedef __attribute__((ext_vector_type(8)))  float  v8f;
typedef __attribute__((ext_vector_type(4)))  float  v4f;
typedef __attribute__((ext_vector_type(4)))  unsigned int u32x4;
typedef __attribute__((ext_vector_type(4)))  int i32x4;
typedef __attribute__((ext_vector_type(8)))  int i32x8;

#define B_   8
#define T_   2048
#define C_   1024
#define H_   16
#define HS_  64
#define G_   32
#define PG_  64
#define ROWS_ 80   // group rows padded: 65 valid (64 + summary), 65..79 zero

#if __has_builtin(__builtin_amdgcn_tensor_load_to_lds)
#define HAVE_TDM 1
#else
#define HAVE_TDM 0
#endif

// ---------------- WMMA helpers (ISA 7.12.2 layouts, wave32) ----------------
__device__ inline v8f zero8() {
  v8f z;
#pragma unroll
  for (int i = 0; i < 8; i++) z[i] = 0.0f;
  return z;
}

__device__ inline v8f wmma_bf16(v16bf a, v16bf b, v8f c) {
  return __builtin_amdgcn_wmma_f32_16x16x32_bf16(false, a, false, b, (short)0, c,
                                                 false, false);
}

// A fragment 16x32 bf16: lane<16 holds row m, K=0..7 & 16..23; lane>=16 K=8..15 & 24..31
__device__ inline v16bf load_frag_a(const bf16_t* base, int ld, int lane) {
  const int m = lane & 15, half = lane >> 4;
  const bf16_t* p = base + m * ld + half * 8;
  v8bf lo = *(const v8bf*)p;
  v8bf hi = *(const v8bf*)(p + 16);
  v16bf r;
#pragma unroll
  for (int i = 0; i < 8; i++) { r[i] = lo[i]; r[i + 8] = hi[i]; }
  return r;
}

// B fragment 32x16 bf16 from row-major [N][K] source: lane<16 holds col n, K=0..7 & 16..23; lane>=16 K=8..15 & 24..31
__device__ inline v16bf load_frag_b(const bf16_t* base, int ld, int lane) {
  const int n = lane & 15, half = lane >> 4;
  const bf16_t* p = base + n * ld + half * 8;
  v8bf lo = *(const v8bf*)p;
  v8bf hi = *(const v8bf*)(p + 16);
  v16bf r;
#pragma unroll
  for (int i = 0; i < 8; i++) { r[i] = lo[i]; r[i + 8] = hi[i]; }
  return r;
}

#if HAVE_TDM
// ---------------- Tensor Data Mover: 2-D bf16 tile global->LDS with row padding ----
// Descriptor packing per CDNA5 ISA.
// pad_interval code: 0=2,1=4,2=8,3=16,4=32 DWORDs; pad_amount code: n => n+1 DWORDs.
__device__ inline void tdm_load_2d_bf16(unsigned lds_off, const void* gaddr,
                                        unsigned tensor_d0, unsigned tensor_d1,
                                        unsigned tile_d0, unsigned tile_d1,
                                        unsigned stride_d0,
                                        unsigned pad_interval, unsigned pad_amount) {
  unsigned long long ga = (unsigned long long)gaddr;
  u32x4 g0;
  g0[0] = 1u;                                            // count=1 (valid), user mode
  g0[1] = lds_off;                                       // lds_addr
  g0[2] = (unsigned)(ga & 0xFFFFFFFFu);                  // global_addr[31:0]
  g0[3] = (unsigned)((ga >> 32) & 0x1FFFFFFu) | (2u << 30);  // addr[56:32], type=2
  i32x8 g1;
  g1[0] = (int)((1u << 16)                               // data_size = 2 bytes
              | (1u << 20)                               // pad_enable
              | (pad_interval << 22) | (pad_amount << 25));
  g1[1] = (int)((tensor_d0 & 0xFFFFu) << 16);            // tensor_dim0[15:0] @ bits 63:48
  g1[2] = (int)((tensor_d0 >> 16) | ((tensor_d1 & 0xFFFFu) << 16));
  g1[3] = (int)((tensor_d1 >> 16) | (tile_d0 << 16));    // tile_dim0 @ bits 127:112
  g1[4] = (int)(tile_d1 & 0xFFFFu);                      // tile_dim1; tile_dim2=0
  g1[5] = (int)stride_d0;                                // tensor_dim0_stride[31:0]
  g1[6] = 0;
  g1[7] = 0;
  i32x4 z4 = {0, 0, 0, 0};
#if defined(__clang_major__) && (__clang_major__ >= 23)
  i32x8 z8 = {0, 0, 0, 0, 0, 0, 0, 0};
  __builtin_amdgcn_tensor_load_to_lds(g0, g1, z4, z4, z8, 0);
#else
  __builtin_amdgcn_tensor_load_to_lds(g0, g1, z4, z4, 0);
#endif
}
#endif  // HAVE_TDM

// ---------------- kernel 1: qkv = x @ Wqkv^T, scatter into group tiles ----------------
__global__ __launch_bounds__(256)
void qkv_gemm_kernel(const float* __restrict__ x, const float* __restrict__ Wqkv,
                     bf16_t* __restrict__ qg, bf16_t* __restrict__ kg,
                     bf16_t* __restrict__ vg) {
  __shared__ bf16_t As[128][40];   // 128 x 32 (+8 pad)
  __shared__ bf16_t Bs[128][40];

  const int tid = threadIdx.x;
  const int lane = tid & 31, wave = tid >> 5;
  const int n0 = blockIdx.x * 128;   // over 3C = 3072
  const int m0 = blockIdx.y * 128;   // over B*T = 16384
  const int wm = wave & 3, wn = wave >> 2;

  v8f acc[2][4];
#pragma unroll
  for (int i = 0; i < 2; i++)
#pragma unroll
    for (int j = 0; j < 4; j++) acc[i][j] = zero8();

  for (int k0 = 0; k0 < 1024; k0 += 32) {
#pragma unroll
    for (int it = 0; it < 4; it++) {
      int chunk = it * 256 + tid;          // 1024 chunks of 4 floats
      int r = chunk >> 3, c = (chunk & 7) * 4;
      v4f fa = *(const v4f*)&x[(size_t)(m0 + r) * 1024 + k0 + c];
      v4f fb = *(const v4f*)&Wqkv[(size_t)(n0 + r) * 1024 + k0 + c];
      v4bf ha, hb;
#pragma unroll
      for (int e = 0; e < 4; e++) { ha[e] = (bf16_t)fa[e]; hb[e] = (bf16_t)fb[e]; }
      *(v4bf*)&As[r][c] = ha;
      *(v4bf*)&Bs[r][c] = hb;
    }
    if (k0 + 32 < 1024) {  // hint next K tile -> global_prefetch_b8
      __builtin_prefetch(&x[(size_t)(m0 + (tid >> 1)) * 1024 + k0 + 32 + (tid & 1) * 16], 0, 0);
      __builtin_prefetch(&Wqkv[(size_t)(n0 + (tid >> 1)) * 1024 + k0 + 32 + (tid & 1) * 16], 0, 0);
    }
    __syncthreads();

    v16bf a0 = load_frag_a(&As[32 * wm][0], 40, lane);
    v16bf a1 = load_frag_a(&As[32 * wm + 16][0], 40, lane);
#pragma unroll
    for (int j = 0; j < 4; j++) {
      v16bf b = load_frag_b(&Bs[64 * wn + 16 * j][0], 40, lane);
      acc[0][j] = wmma_bf16(a0, b, acc[0][j]);
      acc[1][j] = wmma_bf16(a1, b, acc[1][j]);
    }
    __syncthreads();
  }

  // epilogue: scatter into [B,H,G,80,64] bf16 group tiles (rows 0..63)
  const int half = lane >> 4, nn = lane & 15;
#pragma unroll
  for (int i = 0; i < 2; i++)
#pragma unroll
    for (int j = 0; j < 4; j++)
#pragma unroll
      for (int r = 0; r < 8; r++) {
        int gm = m0 + 32 * wm + 16 * i + half * 8 + r;   // b*T + t
        int gn = n0 + 64 * wn + 16 * j + nn;             // col in 3C
        int bb = gm >> 11, t = gm & 2047;
        int which = gn >> 10, cc = gn & 1023;
        int h = cc >> 6, hs = cc & 63;
        int g = t >> 6, p = t & 63;
        bf16_t* dst = which == 0 ? qg : (which == 1 ? kg : vg);
        size_t idx = ((((size_t)bb * H_ + h) * G_ + g) * ROWS_ + p) * HS_ + hs;
        dst[idx] = (bf16_t)acc[i][j][r];
      }
}

// ---------------- kernel 2: summary rows + qs/ks outputs + zero padding ----------------
__global__ __launch_bounds__(64)
void summary_kernel(bf16_t* __restrict__ qg, bf16_t* __restrict__ kg,
                    bf16_t* __restrict__ vg,
                    float* __restrict__ out_qs, float* __restrict__ out_ks) {
  const int blk = blockIdx.x;              // (b*H + h)*G + g
  const int g = blk & 31, h = (blk >> 5) & 15, b = blk >> 9;
  const int hs = threadIdx.x;
  const size_t base = (size_t)blk * ROWS_ * HS_;

#pragma unroll
  for (int w = 0; w < 3; w++) {
    bf16_t* p = (w == 0) ? qg : (w == 1 ? kg : vg);
    float s = 0.0f;
    for (int r = 0; r < 64; r++) s += (float)p[base + (size_t)r * HS_ + hs];
    float mean = s * (1.0f / 64.0f);
    p[base + (size_t)64 * HS_ + hs] = (bf16_t)mean;
    for (int r = 65; r < 80; r++) p[base + (size_t)r * HS_ + hs] = (bf16_t)0.0f;
    if (g < 31 && w < 2) {
      float* o = (w == 0) ? out_qs : out_ks;
      o[(((size_t)b * H_ + h) * 31 + g) * HS_ + hs] = mean;
    }
  }
}

// ---------------- kernel 3: per-group causal attention (L=65, d=64) ----------------
__global__ __launch_bounds__(256)
void attn_kernel(const bf16_t* __restrict__ qg, const bf16_t* __restrict__ kg,
                 const bf16_t* __restrict__ vg,
                 bf16_t* __restrict__ xo, float* __restrict__ xa_sum) {
  __shared__ bf16_t sq[80][72];     // q rows, K=head dim (72 = 64 + 8 pad)
  __shared__ bf16_t sk[80][72];     // k rows
  __shared__ bf16_t svT[64][104];   // v transposed: [hs][key], keys padded to 96
  __shared__ float  S[80][80];      // scores
  __shared__ bf16_t P[80][104];     // probabilities, K padded to 96

  const int blk = blockIdx.x;
  const int g = blk & 31, h = (blk >> 5) & 15, b = blk >> 9;
  const int tid = threadIdx.x, lane = tid & 31, wave = tid >> 5;
  const size_t base = (size_t)blk * ROWS_ * HS_;

#if HAVE_TDM
  // TDM: DMA q,k 80x64 bf16 tiles into LDS with hardware row padding
  // (row = 64 bf16 = 32 DWORDs -> pad_interval code 4; pad = 8 bf16 = 4 DWORDs -> code 3)
  if (wave == 0) {
    tdm_load_2d_bf16((unsigned)(unsigned long long)&sq[0][0], qg + base,
                     64, 80, 64, 80, 64, 4, 3);
    tdm_load_2d_bf16((unsigned)(unsigned long long)&sk[0][0], kg + base,
                     64, 80, 64, 80, 64, 4, 3);
  }
#else
  for (int it = 0; it < 3; it++) {
    int c = it * 256 + tid;                  // 640 chunks of 8 bf16
    if (c < 640) {
      int row = c >> 3, h8 = (c & 7) * 8;
      *(v8bf*)&sq[row][h8] = *(const v8bf*)&qg[base + (size_t)row * HS_ + h8];
      *(v8bf*)&sk[row][h8] = *(const v8bf*)&kg[base + (size_t)row * HS_ + h8];
    }
  }
#endif
  // v staged transposed (manual: TDM cannot transpose)
  for (int it = 0; it < 3; it++) {
    int c = it * 256 + tid;                  // 640 chunks of 8 bf16
    if (c < 640) {
      int row = c >> 3, h8 = (c & 7) * 8;
      v8bf vv = *(const v8bf*)&vg[base + (size_t)row * HS_ + h8];
#pragma unroll
      for (int e = 0; e < 8; e++) svT[h8 + e][row] = vv[e];
    }
  }
#pragma unroll
  for (int it = 0; it < 4; it++) {           // zero svT key cols 80..95
    int c = it * 256 + tid;                  // 64*16 = 1024
    svT[c >> 4][80 + (c & 15)] = (bf16_t)0.0f;
  }
#if HAVE_TDM
  __builtin_amdgcn_s_wait_tensorcnt(0);      // issuing wave drains TENSORcnt
#endif
  __syncthreads();

  // S = (q @ k^T) * scale : 5x5 tiles of 16x16, K=64 (2 wmma steps)
  for (int t = wave; t < 25; t += 8) {
    int ti = t / 5, tj = t % 5;
    v8f acc = zero8();
    acc = wmma_bf16(load_frag_a(&sq[16 * ti][0], 72, lane),
                    load_frag_b(&sk[16 * tj][0], 72, lane), acc);
    acc = wmma_bf16(load_frag_a(&sq[16 * ti][32], 72, lane),
                    load_frag_b(&sk[16 * tj][32], 72, lane), acc);
    const int half = lane >> 4, nn = lane & 15;
#pragma unroll
    for (int r = 0; r < 8; r++)
      S[16 * ti + half * 8 + r][16 * tj + nn] = acc[r] * 0.125f;
  }
  __syncthreads();

  // masked softmax rows 0..64 (row r attends cols 0..r); write P (bf16, zero-padded)
  if (tid < 65) {
    const int row = tid;
    float m = -3.0e38f;
    for (int c = 0; c <= row; c++) m = fmaxf(m, S[row][c]);
    float sum = 0.0f;
    for (int c = 0; c <= row; c++) sum += __expf(S[row][c] - m);
    float inv = 1.0f / sum;
    for (int c = 0; c < 96; c++)
      P[row][c] = (c <= row) ? (bf16_t)(__expf(S[row][c] - m) * inv) : (bf16_t)0.0f;
  } else if (tid < 80) {
    const int row = tid;                     // 65..79 pad rows
    for (int c = 0; c < 96; c++) P[row][c] = (bf16_t)0.0f;
  }
  __syncthreads();

  // O = P @ V : 5x4 tiles, K=96 (3 wmma steps)
  for (int t = wave; t < 20; t += 8) {
    int ti = t >> 2, tj = t & 3;
    v8f acc = zero8();
#pragma unroll
    for (int ks = 0; ks < 3; ks++)
      acc = wmma_bf16(load_frag_a(&P[16 * ti][32 * ks], 104, lane),
                      load_frag_b(&svT[16 * tj][32 * ks], 104, lane), acc);
    const int half = lane >> 4, nn = lane & 15;
#pragma unroll
    for (int r = 0; r < 8; r++) {
      int row = 16 * ti + half * 8 + r;
      int hs = 16 * tj + nn;
      float val = acc[r];
      if (row < 64) {
        xo[((size_t)(b * T_ + g * 64 + row)) * C_ + h * 64 + hs] = (bf16_t)val;
      } else if (row == 64) {
        xa_sum[(size_t)blk * HS_ + hs] = val;   // summary-row output (fp32)
      }
    }
  }
}

// ---------------- kernel 4: out = xo @ Wproj^T (fp32 epilogue) ----------------
__global__ __launch_bounds__(256)
void proj_gemm_kernel(const bf16_t* __restrict__ xo, const float* __restrict__ Wp,
                      float* __restrict__ out) {
  __shared__ bf16_t As[128][40];
  __shared__ bf16_t Bs[128][40];

  const int tid = threadIdx.x;
  const int lane = tid & 31, wave = tid >> 5;
  const int n0 = blockIdx.x * 128;   // over C = 1024
  const int m0 = blockIdx.y * 128;   // over B*T = 16384
  const int wm = wave & 3, wn = wave >> 2;

  v8f acc[2][4];
#pragma unroll
  for (int i = 0; i < 2; i++)
#pragma unroll
    for (int j = 0; j < 4; j++) acc[i][j] = zero8();

  for (int k0 = 0; k0 < 1024; k0 += 32) {
#if HAVE_TDM
    // TDM: DMA A tile (128x32 bf16, already bf16 in ws) with hw padding to stride 40
    // (row = 32 bf16 = 16 DWORDs -> pad_interval code 3; pad = 4 DWORDs -> code 3)
    if (wave == 0) {
      tdm_load_2d_bf16((unsigned)(unsigned long long)&As[0][0],
                       &xo[(size_t)m0 * 1024 + k0],
                       32, 128, 32, 128, 1024, 3, 3);
    }
#else
#pragma unroll
    for (int it = 0; it < 2; it++) {         // A: bf16 direct, 512 chunks of 8
      int c = it * 256 + tid;
      int r = c >> 2, cc = (c & 3) * 8;
      *(v8bf*)&As[r][cc] = *(const v8bf*)&xo[(size_t)(m0 + r) * 1024 + k0 + cc];
    }
#endif
#pragma unroll
    for (int it = 0; it < 4; it++) {         // B: fp32 -> bf16
      int chunk = it * 256 + tid;
      int r = chunk >> 3, c = (chunk & 7) * 4;
      v4f fb = *(const v4f*)&Wp[(size_t)(n0 + r) * 1024 + k0 + c];
      v4bf hb;
#pragma unroll
      for (int e = 0; e < 4; e++) hb[e] = (bf16_t)fb[e];
      *(v4bf*)&Bs[r][c] = hb;
    }
    if (k0 + 32 < 1024)
      __builtin_prefetch(&Wp[(size_t)(n0 + (tid >> 1)) * 1024 + k0 + 32 + (tid & 1) * 16], 0, 0);
#if HAVE_TDM
    __builtin_amdgcn_s_wait_tensorcnt(0);
#endif
    __syncthreads();

    v16bf a0 = load_frag_a(&As[32 * wm][0], 40, lane);
    v16bf a1 = load_frag_a(&As[32 * wm + 16][0], 40, lane);
#pragma unroll
    for (int j = 0; j < 4; j++) {
      v16bf b = load_frag_b(&Bs[64 * wn + 16 * j][0], 40, lane);
      acc[0][j] = wmma_bf16(a0, b, acc[0][j]);
      acc[1][j] = wmma_bf16(a1, b, acc[1][j]);
    }
    __syncthreads();
  }

  const int half = lane >> 4, nn = lane & 15;
#pragma unroll
  for (int i = 0; i < 2; i++)
#pragma unroll
    for (int j = 0; j < 4; j++)
#pragma unroll
      for (int r = 0; r < 8; r++) {
        int gm = m0 + 32 * wm + 16 * i + half * 8 + r;
        int gn = n0 + 64 * wn + 16 * j + nn;
        out[(size_t)gm * 1024 + gn] = acc[i][j][r];
      }
}

// ---------------- kernel 5: vs = causal_attn(qs, ks, xa_sum[:,:,:-1]) ----------------
__global__ __launch_bounds__(256)
void vs_kernel(const float* __restrict__ qs, const float* __restrict__ ks,
               const float* __restrict__ xa_sum, float* __restrict__ vs) {
  __shared__ float Q[31][64], K[31][64], V[31][64];
  __shared__ float Sc[31][32];
  const int bh = blockIdx.x;                 // b*H + h
  const int tid = threadIdx.x;

  for (int it = 0; it < 8; it++) {
    int c = it * 256 + tid;                  // 31*64 = 1984
    if (c < 1984) {
      int i = c >> 6, hs = c & 63;
      size_t idx = ((size_t)bh * 31 + i) * 64 + hs;
      Q[i][hs] = qs[idx];
      K[i][hs] = ks[idx];
      V[i][hs] = xa_sum[((size_t)bh * 32 + i) * 64 + hs];  // groups 0..30
    }
  }
  __syncthreads();

  for (int it = 0; it < 4; it++) {
    int c = it * 256 + tid;                  // 31*31 = 961
    if (c < 961) {
      int i = c / 31, j = c % 31;
      if (j <= i) {
        float s = 0.0f;
        for (int d = 0; d < 64; d++) s += Q[i][d] * K[j][d];
        Sc[i][j] = s * 0.125f;
      } else {
        Sc[i][j] = -3.0e38f;
      }
    }
  }
  __syncthreads();

  if (tid < 31) {
    const int i = tid;
    float m = -3.0e38f;
    for (int j = 0; j <= i; j++) m = fmaxf(m, Sc[i][j]);
    float sum = 0.0f;
    for (int j = 0; j <= i; j++) { float e = __expf(Sc[i][j] - m); Sc[i][j] = e; sum += e; }
    float inv = 1.0f / sum;
    for (int j = 0; j <= i; j++) Sc[i][j] *= inv;
    for (int j = i + 1; j < 31; j++) Sc[i][j] = 0.0f;
  }
  __syncthreads();

  for (int it = 0; it < 8; it++) {
    int c = it * 256 + tid;
    if (c < 1984) {
      int i = c >> 6, hs = c & 63;
      float s = 0.0f;
      for (int j = 0; j <= i; j++) s += Sc[i][j] * V[j][hs];
      vs[((size_t)bh * 31 + i) * 64 + hs] = s;
    }
  }
}

// ---------------- launch ----------------
extern "C" void kernel_launch(void* const* d_in, const int* in_sizes, int n_in,
                              void* d_out, int out_size, void* d_ws, size_t ws_size,
                              hipStream_t stream) {
  const float* x     = (const float*)d_in[0];
  const float* Wqkv  = (const float*)d_in[1];
  const float* Wproj = (const float*)d_in[2];

  float* out    = (float*)d_out;                           // [B,T,C]
  float* out_qs = out + (size_t)B_ * T_ * C_;              // [B,H,31,64]
  float* out_ks = out_qs + (size_t)B_ * H_ * 31 * 64;
  float* out_vs = out_ks + (size_t)B_ * H_ * 31 * 64;

  char* ws = (char*)d_ws;
  const size_t gsz = (size_t)B_ * H_ * G_ * ROWS_ * HS_ * sizeof(bf16_t);  // 40 MiB
  bf16_t* qg = (bf16_t*)(ws);
  bf16_t* kg = (bf16_t*)(ws + gsz);
  bf16_t* vg = (bf16_t*)(ws + 2 * gsz);
  bf16_t* xo = (bf16_t*)(ws + 3 * gsz);                    // [B*T, C] bf16
  float* xa_sum = (float*)(ws + 3 * gsz + (size_t)B_ * T_ * C_ * sizeof(bf16_t)); // [B,H,G,64]

  qkv_gemm_kernel<<<dim3(24, 128), 256, 0, stream>>>(x, Wqkv, qg, kg, vg);
  summary_kernel<<<B_ * H_ * G_, 64, 0, stream>>>(qg, kg, vg, out_qs, out_ks);
  attn_kernel<<<B_ * H_ * G_, 256, 0, stream>>>(qg, kg, vg, xo, xa_sum);
  proj_gemm_kernel<<<dim3(8, 128), 256, 0, stream>>>(xo, Wproj, out);
  vs_kernel<<<B_ * H_, 256, 0, stream>>>(out_qs, out_ks, xa_sum, out_vs);
}
